// FlashMoE_13692355739734
// MI455X (gfx1250) — matrix operand order repaired and checked by reading, compiled, not run on hardware
//
#include <hip/hip_runtime.h>
#include <hip/hip_bf16.h>

// ---------------------------------------------------------------------------
// Problem constants (from reference)
// ---------------------------------------------------------------------------
#define NTOK 4096   // B*S
#define HDIM 2048   // hidden
#define IDIM 8192   // intermediate
#define NKEY 64
#define TOPK 16

// LDS tile geometry: K-tile = 32 bf16 per row, row padded to 40 ushorts (80B)
// -> 16B aligned rows, and (row*20 mod 64) banks distinct for 16 rows.
#define ROWP 40

typedef __attribute__((ext_vector_type(16))) __bf16 v16bf;
typedef __attribute__((ext_vector_type(8)))  float  v8f;
typedef __attribute__((ext_vector_type(4)))  unsigned int v4u;
typedef __attribute__((ext_vector_type(2)))  unsigned int v2u;
typedef __attribute__((ext_vector_type(4)))  float  v4f;

// ---------------------------------------------------------------------------
// helpers
// ---------------------------------------------------------------------------
__device__ __forceinline__ unsigned short f2bf(float f) {
    unsigned int u = __float_as_uint(f);
    u += 0x7FFFu + ((u >> 16) & 1u);   // round-to-nearest-even
    return (unsigned short)(u >> 16);
}

__device__ __forceinline__ v8f v8f_zero() {
    v8f v;
#pragma unroll
    for (int i = 0; i < 8; ++i) v[i] = 0.0f;
    return v;
}

// gfx1250 async global->LDS copy, 16B per lane.  VDST = LDS offset (wave
// allocation relative = low 32 bits of the generic pointer), VADDR = 64-bit
// global address.  Tracked by ASYNCcnt.
__device__ __forceinline__ void async_ld16(unsigned lds_off, const void* gptr) {
    asm volatile("global_load_async_to_lds_b128 %0, %1, off"
                 :: "v"(lds_off), "v"((unsigned long long)(size_t)gptr)
                 : "memory");
}
__device__ __forceinline__ void wait_async0() {
    asm volatile("s_wait_asynccnt 0x0" ::: "memory");
}

// A-matrix fragment (16x32 bf16, M on lanes) from an LDS tile (ROWP-padded):
//   lane L: M = L%16, half = L/16
//   vec elems 0..7  : K = half*8 + 0..7 ;  elems 8..15: K = 16 + half*8 + 0..7
__device__ __forceinline__ v16bf lds_frag_a(const unsigned short* __restrict__ t,
                                            int row, int lane) {
    int r = lane & 15, h = lane >> 4;
    const unsigned short* p = t + (row + r) * ROWP + h * 8;
    union { v16bf v; v4u q[2]; } u;
    u.q[0] = *(const v4u*)p;
    u.q[1] = *(const v4u*)(p + 16);
    return u.v;
}

// B-matrix fragment (32x16 bf16, N on lanes) from an LDS tile:
//   lane L: N = L%16, half = L/16 ; elems 0..7: K = half*8 + 0..7 ; elems 8..15: K = 16 + half*8 + 0..7
__device__ __forceinline__ v16bf lds_frag_b(const unsigned short* __restrict__ t,
                                            int col, int lane) {
    int c = lane & 15, h = lane >> 4;
    const unsigned short* p = t + (col + c) * ROWP + h * 8;
    union { v16bf v; v4u q[2]; } u;
    u.q[0] = *(const v4u*)p;
    u.q[1] = *(const v4u*)(p + 16);
    return u.v;
}

__device__ __forceinline__ v8f wmma_bf16(v16bf a, v16bf b, v8f c) {
    return __builtin_amdgcn_wmma_f32_16x16x32_bf16(false, a, false, b,
                                                   (short)0, c, false, false);
}

// ---------------------------------------------------------------------------
// fp32 -> bf16 bulk convert (4 elems / thread)
// ---------------------------------------------------------------------------
__global__ __launch_bounds__(256) void k_f2bf(const float* __restrict__ src,
                                              unsigned short* __restrict__ dst,
                                              int n4) {
    int i = blockIdx.x * blockDim.x + threadIdx.x;
    if (i >= n4) return;
    v4f v = ((const v4f*)src)[i];
    v2u o;
    o.x = (unsigned)f2bf(v.x) | ((unsigned)f2bf(v.y) << 16);
    o.y = (unsigned)f2bf(v.z) | ((unsigned)f2bf(v.w) << 16);
    ((v2u*)dst)[i] = o;
}

// ---------------------------------------------------------------------------
// BatchNorm statistics: per-feature mean / rstd over N tokens (biased var)
// ---------------------------------------------------------------------------
__global__ __launch_bounds__(256) void k_bnstats(const float* __restrict__ x,
                                                 float* __restrict__ mu,
                                                 float* __restrict__ rstd) {
    int h = blockIdx.x * blockDim.x + threadIdx.x;
    if (h >= HDIM) return;
    float s = 0.f, ss = 0.f;
    for (int n = 0; n < NTOK; ++n) {
        float v = x[(size_t)n * HDIM + h];
        s += v; ss += v * v;
    }
    float m = s * (1.0f / NTOK);
    float var = ss * (1.0f / NTOK) - m * m;
    mu[h] = m;
    rstd[h] = rsqrtf(var + 1e-5f);
}

// ---------------------------------------------------------------------------
// Router: logits, log-softmax, exact top-16 of the 64x64 sum grid.
// One block (128 thr) per token.
// ---------------------------------------------------------------------------
__global__ __launch_bounds__(128) void k_router(const float* __restrict__ x,
                                                const float* __restrict__ rgw,
                                                const float* __restrict__ mu,
                                                const float* __restrict__ rstd,
                                                float* __restrict__ rlog_out, // [2,N,64]
                                                int* __restrict__ idx,        // [N,16]
                                                float* __restrict__ rw) {     // [N,16]
    __shared__ float xn[HDIM];
    __shared__ float lg[128];
    __shared__ float lp[128];
    int n = blockIdx.x, t = threadIdx.x;

    for (int h = t; h < HDIM; h += 128)
        xn[h] = (x[(size_t)n * HDIM + h] - mu[h]) * rstd[h];
    __syncthreads();

    const float* wr = rgw + (size_t)t * HDIM;
    float acc = 0.f;
    for (int h = 0; h < HDIM; ++h) acc = fmaf(xn[h], wr[h], acc);
    lg[t] = acc;
    int c = t >> 6, j = t & 63;
    rlog_out[(size_t)c * (NTOK * NKEY) + (size_t)n * NKEY + j] = acc;
    __syncthreads();

    float mx = -1e30f;
    for (int q = 0; q < NKEY; ++q) mx = fmaxf(mx, lg[c * NKEY + q]);
    float se = 0.f;
    for (int q = 0; q < NKEY; ++q) se += __expf(lg[c * NKEY + q] - mx);
    lp[t] = acc - mx - __logf(se);
    __syncthreads();

    if (t == 0) {
        float v0[16], v1[16];
        int   i0[16], i1[16];
        unsigned long long taken = 0ull;
        for (int s = 0; s < TOPK; ++s) {
            float best = -1e30f; int bi = 0;
            for (int q = 0; q < NKEY; ++q)
                if (!((taken >> q) & 1ull) && lp[q] > best) { best = lp[q]; bi = q; }
            taken |= 1ull << bi; v0[s] = best; i0[s] = bi;
        }
        taken = 0ull;
        for (int s = 0; s < TOPK; ++s) {
            float best = -1e30f; int bi = 0;
            for (int q = 0; q < NKEY; ++q)
                if (!((taken >> q) & 1ull) && lp[NKEY + q] > best) { best = lp[NKEY + q]; bi = q; }
            taken |= 1ull << bi; v1[s] = best; i1[s] = bi;
        }
        float rv[16]; int ri[16];
        for (int s = 0; s < TOPK; ++s) { rv[s] = -1e30f; ri[s] = 0; }
        for (int a = 0; a < TOPK; ++a)
            for (int b = 0; b < TOPK; ++b) {
                float sc = v0[a] + v1[b];
                if (sc > rv[TOPK - 1]) {
                    int pos = TOPK - 1;
                    while (pos > 0 && rv[pos - 1] < sc) {
                        rv[pos] = rv[pos - 1]; ri[pos] = ri[pos - 1]; --pos;
                    }
                    rv[pos] = sc; ri[pos] = i0[a] * NKEY + i1[b];
                }
            }
        for (int s = 0; s < TOPK; ++s) {
            idx[n * TOPK + s] = ri[s];
            rw[n * TOPK + s] = __expf(rv[s]);
        }
    }
}

// ---------------------------------------------------------------------------
// PEER expert: gather down/up rows (L2-resident), dot + weighted sum.
// One block (256 thr = 8 waves) per token; wave32 butterfly + 8-slot combine.
// ---------------------------------------------------------------------------
__global__ __launch_bounds__(256) void k_expert(const float* __restrict__ x,
                                                const float* __restrict__ de,
                                                const float* __restrict__ ue,
                                                const int* __restrict__ idx,
                                                const float* __restrict__ rw,
                                                float* __restrict__ eo) {
    __shared__ float red[8];
    int n = blockIdx.x, t = threadIdx.x;
    int lane = t & 31, w = t >> 5;
    float xr[8], acc[8];
#pragma unroll
    for (int j = 0; j < 8; ++j) {
        xr[j] = x[(size_t)n * HDIM + t + j * 256];
        acc[j] = 0.f;
    }
    for (int k = 0; k < TOPK; ++k) {
        int e = idx[n * TOPK + k];
        float wgt = rw[n * TOPK + k];
        const float* dp = de + (size_t)e * HDIM;
        float p = 0.f;
#pragma unroll
        for (int j = 0; j < 8; ++j) p = fmaf(xr[j], dp[t + j * 256], p);
        // wave32 butterfly reduction (lane-permute HW, no LDS)
#pragma unroll
        for (int m = 16; m > 0; m >>= 1) p += __shfl_xor(p, m, 32);
        if (lane == 0) red[w] = p;
        __syncthreads();
        float hk = 0.f;
#pragma unroll
        for (int q = 0; q < 8; ++q) hk += red[q];
        __syncthreads();                 // red[] reusable next iteration
        float sc = wgt * hk;
        const float* up = ue + (size_t)e * HDIM;
#pragma unroll
        for (int j = 0; j < 8; ++j) acc[j] = fmaf(sc, up[t + j * 256], acc[j]);
    }
#pragma unroll
    for (int j = 0; j < 8; ++j) eo[(size_t)n * HDIM + t + j * 256] = acc[j];
}

// ---------------------------------------------------------------------------
// Fused gate/up GEMM + SiLU:  Abf16[m,i] = silu(x@gw.T) * (x@uw.T)
// block tile 128(M) x 128(I), 8 waves (2x4), wave tile 64x32, shared A frags.
// Async double-buffered LDS staging, one barrier per K-step, peeled last step.
// ---------------------------------------------------------------------------
__global__ __launch_bounds__(256, 1) void k_gateup(const unsigned short* __restrict__ xb,
                                                   const unsigned short* __restrict__ gwb,
                                                   const unsigned short* __restrict__ uwb,
                                                   unsigned short* __restrict__ Ab) {
    __shared__ __align__(16) unsigned short smA[2][128 * ROWP];
    __shared__ __align__(16) unsigned short smG[2][128 * ROWP];
    __shared__ __align__(16) unsigned short smU[2][128 * ROWP];

    int tid = threadIdx.x;
    int lane = tid & 31;
    int w = tid >> 5;
    int wm = w >> 2, wn = w & 3;
    int m0b = blockIdx.y * 128;
    int n0b = blockIdx.x * 128;

    auto stage = [&](int buf, int k) {
#pragma unroll
        for (int j = 0; j < 2; ++j) {
            int c = tid + 256 * j;          // 512 16B-chunks per 128x32 tile
            int row = c >> 2, q = c & 3;
            unsigned lo = (unsigned)(row * ROWP + q * 8) * 2u;
            size_t go = (size_t)row * HDIM + k + q * 8;
            async_ld16((unsigned)(size_t)&smA[buf][0] + lo, xb  + (size_t)(m0b)*HDIM + go);
            async_ld16((unsigned)(size_t)&smG[buf][0] + lo, gwb + (size_t)(n0b)*HDIM + go);
            async_ld16((unsigned)(size_t)&smU[buf][0] + lo, uwb + (size_t)(n0b)*HDIM + go);
        }
    };

    v8f ag[8], au[8];
#pragma unroll
    for (int f = 0; f < 8; ++f) { ag[f] = v8f_zero(); au[f] = v8f_zero(); }

    auto compute = [&](int buf) {
        v16bf a[4];
#pragma unroll
        for (int i = 0; i < 4; ++i)
            a[i] = lds_frag_a(smA[buf], wm * 64 + 16 * i, lane);
        v16bf bg[2], bu[2];
#pragma unroll
        for (int j = 0; j < 2; ++j) {
            bg[j] = lds_frag_b(smG[buf], wn * 32 + 16 * j, lane);
            bu[j] = lds_frag_b(smU[buf], wn * 32 + 16 * j, lane);
        }
#pragma unroll
        for (int i = 0; i < 4; ++i)
#pragma unroll
            for (int j = 0; j < 2; ++j) {
                ag[i * 2 + j] = wmma_bf16(a[i], bg[j], ag[i * 2 + j]);
                au[i * 2 + j] = wmma_bf16(a[i], bu[j], au[i * 2 + j]);
            }
    };

    stage(0, 0);
    int buf = 0;
    for (int k = 0; k < HDIM - 32; k += 32, buf ^= 1) {
        wait_async0();      // current stage landed in LDS (this wave)
        __syncthreads();    // all waves staged; prev buffer's readers done
        stage(buf ^ 1, k + 32);
        compute(buf);
    }
    wait_async0();
    __syncthreads();
    compute(buf);           // peeled last K-step

#pragma unroll
    for (int i = 0; i < 4; ++i)
#pragma unroll
        for (int j = 0; j < 2; ++j) {
            int f = i * 2 + j;
#pragma unroll
            for (int r = 0; r < 8; ++r) {
                float g = ag[f][r], uu = au[f][r];
                float act = (g / (1.0f + __expf(-g))) * uu;   // silu(g) * u
                int m = m0b + wm * 64 + 16 * i + (lane >> 4) * 8 + r;
                int nn = n0b + wn * 32 + 16 * j + (lane & 15);
                Ab[(size_t)m * IDIM + nn] = f2bf(act);
            }
        }
}

// ---------------------------------------------------------------------------
// Down GEMM + expert add:  out[m,h] = Abf16 @ down_w.T + expert_out
// block tile 128(M) x 256(H), 8 waves (2x4), wave tile 64x64.
// Async double-buffered LDS staging, one barrier per K-step, peeled last step.
// ---------------------------------------------------------------------------
__global__ __launch_bounds__(256, 1) void k_down(const unsigned short* __restrict__ Ab,
                                                 const unsigned short* __restrict__ dwb,
                                                 const float* __restrict__ eo,
                                                 float* __restrict__ out) {
    __shared__ __align__(16) unsigned short smA[2][128 * ROWP];
    __shared__ __align__(16) unsigned short smB[2][256 * ROWP];

    int tid = threadIdx.x;
    int lane = tid & 31;
    int w = tid >> 5;
    int wm = w >> 2, wn = w & 3;
    int m0b = blockIdx.y * 128;
    int n0b = blockIdx.x * 256;

    auto stage = [&](int buf, int k) {
#pragma unroll
        for (int j = 0; j < 2; ++j) {       // A: 512 chunks
            int c = tid + 256 * j;
            int row = c >> 2, q = c & 3;
            async_ld16((unsigned)(size_t)&smA[buf][0] + (unsigned)(row * ROWP + q * 8) * 2u,
                       Ab + (size_t)(m0b + row) * IDIM + k + q * 8);
        }
#pragma unroll
        for (int j = 0; j < 4; ++j) {       // B: 1024 chunks
            int c = tid + 256 * j;
            int row = c >> 2, q = c & 3;
            async_ld16((unsigned)(size_t)&smB[buf][0] + (unsigned)(row * ROWP + q * 8) * 2u,
                       dwb + (size_t)(n0b + row) * IDIM + k + q * 8);
        }
    };

    v8f acc[16];
#pragma unroll
    for (int f = 0; f < 16; ++f) acc[f] = v8f_zero();

    auto compute = [&](int buf) {
        v16bf a[4];
#pragma unroll
        for (int i = 0; i < 4; ++i)
            a[i] = lds_frag_a(smA[buf], wm * 64 + 16 * i, lane);
        v16bf b[4];
#pragma unroll
        for (int j = 0; j < 4; ++j)
            b[j] = lds_frag_b(smB[buf], wn * 64 + 16 * j, lane);
#pragma unroll
        for (int i = 0; i < 4; ++i)
#pragma unroll
            for (int j = 0; j < 4; ++j)
                acc[i * 4 + j] = wmma_bf16(a[i], b[j], acc[i * 4 + j]);
    };

    stage(0, 0);
    int buf = 0;
    for (int k = 0; k < IDIM - 32; k += 32, buf ^= 1) {
        wait_async0();
        __syncthreads();
        stage(buf ^ 1, k + 32);
        compute(buf);
    }
    wait_async0();
    __syncthreads();
    compute(buf);

#pragma unroll
    for (int i = 0; i < 4; ++i)
#pragma unroll
        for (int j = 0; j < 4; ++j) {
            int f = i * 4 + j;
#pragma unroll
            for (int r = 0; r < 8; ++r) {
                int m = m0b + wm * 64 + 16 * i + (lane >> 4) * 8 + r;
                int nn = n0b + wn * 64 + 16 * j + (lane & 15);
                size_t o = (size_t)m * HDIM + nn;
                out[o] = acc[f][r] + eo[o];
            }
        }
}

// ---------------------------------------------------------------------------
// Launcher
// ---------------------------------------------------------------------------
extern "C" void kernel_launch(void* const* d_in, const int* in_sizes, int n_in,
                              void* d_out, int out_size, void* d_ws, size_t ws_size,
                              hipStream_t stream) {
    (void)in_sizes; (void)n_in; (void)out_size; (void)ws_size;

    const float* x   = (const float*)d_in[0];   // [N, H]
    const float* rgw = (const float*)d_in[1];   // [128, H]
    const float* de  = (const float*)d_in[2];   // [4096, H]
    const float* ue  = (const float*)d_in[3];   // [4096, H]
    const float* gw  = (const float*)d_in[4];   // [I, H]
    const float* uw  = (const float*)d_in[5];   // [I, H]
    const float* dw  = (const float*)d_in[6];   // [H, I]

    float* out      = (float*)d_out;             // [N, H]
    float* rlog_out = out + (size_t)NTOK * HDIM; // [2, N, 64]

    char* base = (char*)d_ws;
    size_t o = 0;
    auto carve = [&](size_t bytes) {
        void* p = base + o;
        o += (bytes + 255) & ~(size_t)255;
        return p;
    };
    float*          mu   = (float*)carve((size_t)HDIM * 4);
    float*          rstd = (float*)carve((size_t)HDIM * 4);
    int*            idx  = (int*)carve((size_t)NTOK * TOPK * 4);
    float*          rw   = (float*)carve((size_t)NTOK * TOPK * 4);
    float*          eo   = (float*)carve((size_t)NTOK * HDIM * 4);
    unsigned short* xb   = (unsigned short*)carve((size_t)NTOK * HDIM * 2);
    unsigned short* gwb  = (unsigned short*)carve((size_t)IDIM * HDIM * 2);
    unsigned short* uwb  = (unsigned short*)carve((size_t)IDIM * HDIM * 2);
    unsigned short* dwb  = (unsigned short*)carve((size_t)HDIM * IDIM * 2);
    unsigned short* Ab   = (unsigned short*)carve((size_t)NTOK * IDIM * 2);

    {
        int n4 = NTOK * HDIM / 4;
        k_f2bf<<<n4 / 256, 256, 0, stream>>>(x, xb, n4);
        n4 = IDIM * HDIM / 4;
        k_f2bf<<<n4 / 256, 256, 0, stream>>>(gw, gwb, n4);
        k_f2bf<<<n4 / 256, 256, 0, stream>>>(uw, uwb, n4);
        k_f2bf<<<n4 / 256, 256, 0, stream>>>(dw, dwb, n4);
    }
    k_bnstats<<<HDIM / 256, 256, 0, stream>>>(x, mu, rstd);
    k_router<<<NTOK, 128, 0, stream>>>(x, rgw, mu, rstd, rlog_out, idx, rw);
    k_expert<<<NTOK, 256, 0, stream>>>(x, de, ue, idx, rw, eo);
    k_gateup<<<dim3(IDIM / 128, NTOK / 128), 256, 0, stream>>>(xb, gwb, uwb, Ab);
    k_down<<<dim3(HDIM / 256, NTOK / 128), 256, 0, stream>>>(Ab, dwb, eo, out);
}
